// DirectedDualPNA_30339648979104
// MI455X (gfx1250) — matrix operand order repaired and checked by reading, not compiled, not run on hardware
//
#include <hip/hip_runtime.h>
#include <hip/hip_bf16.h>
#include <math.h>

// ---------------------------------------------------------------------------
// DirectedDualPNA forward for MI455X (gfx1250): WMMA f16->f32 GEMMs with
// double-buffered LDS-staged A tiles + transposed f16 weights, factored PNA
// message passing with on-chip (L2) float atomics.
// ---------------------------------------------------------------------------

typedef _Float16 v16h __attribute__((ext_vector_type(16)));
typedef _Float16 v8h  __attribute__((ext_vector_type(8)));
typedef _Float16 v4h  __attribute__((ext_vector_type(4)));
typedef float    v8f  __attribute__((ext_vector_type(8)));

#define NNODES 50000
#define NEDGES 800000
#define FDIM   128
#define NLAYER 2
#define LDSROW 40               // padded LDS row stride (f16) to spread banks

// ------------------------- generic WMMA GEMM -------------------------------
// C[M x Nout] (f32, ldc) = base + rowscale(m) * (A[M x K](f16,lda) @ B)
//   B given transposed+f16: Bt[n][kfull], B[k][n] = Bt[n*ldk + r0 + k]
//   base = accumulate ? C_prev : (bias ? bias[n] : 0)
// Block = 4 waves: one 16-row tile (blockIdx.x), 4 consecutive 16-col tiles.
// A tile (16x32 f16) double-buffered through LDS; next tile's global load is
// issued before the current WMMA so its latency hides behind compute.
// Out-of-range waves recompute the last column tile (no EXEC manipulation
// around the WMMA); only the store is guarded.
__global__ void wmma_gemm_kernel(const _Float16* __restrict__ A, int lda,
                                 const _Float16* __restrict__ Bt, int ldk, int r0,
                                 float* __restrict__ C, int ldc,
                                 const float* __restrict__ bias,
                                 const float* __restrict__ rowscale,
                                 int K, int Nout, int tilesN, int accumulate)
{
    __shared__ _Float16 As[2][16 * LDSROW];

    const int lane = threadIdx.x & 31;
    const int wave = threadIdx.x >> 5;
    const int tm   = blockIdx.x;
    const int tn   = blockIdx.y * 4 + wave;
    const int tnc  = (tn < tilesN) ? tn : (tilesN - 1);   // clamp, no branches

    const int h    = lane >> 4;                 // half-select (ISA 7.12.2)
    const int l15  = lane & 15;
    const int ncol = tnc * 16 + l15;            // B/C column owned by this lane

    // cooperative A-tile loader mapping: 128 thr x 8B = 16 rows x 64B
    const int trow  = threadIdx.x >> 3;         // 0..15
    const int tsegh = (threadIdx.x & 7) * 4;    // f16 offset 0..28
    const _Float16* aG = A + (size_t)(tm * 16 + trow) * lda + tsegh;
    const _Float16* bG = Bt + (size_t)ncol * ldk + r0 + 16 * h;
    const int sWr = trow * LDSROW + tsegh;      // LDS write slot (f16 units)
    const int sRd = l15 * LDSROW + 16 * h;      // LDS read base  (f16 units)

    v8f acc = {};

    // prologue: stage tile 0
    *(v4h*)(&As[0][sWr]) = *(const v4h*)(aG);
    __syncthreads();

    int buf = 0;
    for (int k0 = 0; k0 < K; k0 += 32) {
        const bool more = (k0 + 32 < K);
        // issue next A chunk early (global latency overlaps current compute)
        v4h anext;
        if (more) anext = *(const v4h*)(aG + k0 + 32);

        // A fragment from LDS: lanes 0-15 hold K[0..15] of the tile,
        //                      lanes 16-31 hold K[16..31]
        v8h c0 = *(const v8h*)(&As[buf][sRd]);
        v8h c1 = *(const v8h*)(&As[buf][sRd + 8]);
        v16h a;
#pragma unroll
        for (int i = 0; i < 8; ++i) { a[i] = c0[i]; a[8 + i] = c1[i]; }

        // B fragment: element e = B[k0+16h+e][ncol] = contiguous Bt row (32B)
        v16h b = *(const v16h*)(bG + k0);

        acc = __builtin_amdgcn_wmma_f32_16x16x32_f16(
                  /*neg_a=*/false, a, /*neg_b=*/false, b,
                  /*c_mod=*/(short)0, acc,
                  /*reuse_a=*/false, /*reuse_b=*/false);

        if (more) {
            *(v4h*)(&As[buf ^ 1][sWr]) = anext;
            __syncthreads();            // one barrier per K-step
            buf ^= 1;
        }
    }

    // ---- epilogue: C/D layout = row (16*tm + v + 8h), col ncol
    if (tn < tilesN && (tn * 16 + l15) < Nout) {
#pragma unroll
        for (int v = 0; v < 8; ++v) {
            int row = tm * 16 + v + 8 * h;
            size_t ci = (size_t)row * ldc + ncol;
            float base = accumulate ? C[ci] : (bias ? bias[ncol] : 0.0f);
            float rs   = rowscale ? rowscale[row] : 1.0f;
            C[ci] = base + rs * acc[v];
        }
    }
}

// ---------------- weight transpose + f32->f16 conversion -------------------
// W[Kf x Nc] (f32, row-major) -> Bt[Ncpad x Kf] (f16), zero-padded columns.
__global__ void wtrans_f16_kernel(const float* __restrict__ W,
                                  _Float16* __restrict__ Bt,
                                  int Kf, int Nc, int Ncpad)
{
    int i = blockIdx.x * blockDim.x + threadIdx.x;
    if (i >= Ncpad * Kf) return;
    int n = i / Kf, k = i - n * Kf;
    Bt[i] = (n < Nc) ? (_Float16)W[(size_t)k * Nc + n] : (_Float16)0.0f;
}

// ------------------------- elementwise helpers -----------------------------
__global__ void f32_to_f16_kernel(const float* __restrict__ in,
                                  _Float16* __restrict__ out, int n)
{
    int i = blockIdx.x * blockDim.x + threadIdx.x;
    if (i < n) out[i] = (_Float16)in[i];
}

__global__ void relu_f32_to_f16_kernel(const float* __restrict__ in,
                                       _Float16* __restrict__ out, int n)
{
    int i = blockIdx.x * blockDim.x + threadIdx.x;
    if (i < n) out[i] = (_Float16)fmaxf(in[i], 0.0f);
}

__global__ void init_agg_kernel(float* __restrict__ sum, float* __restrict__ sumsq,
                                float* __restrict__ mn, float* __restrict__ mx,
                                float* __restrict__ cnt, int nn)
{
    int i = blockIdx.x * blockDim.x + threadIdx.x;
    if (i < nn * FDIM) {
        sum[i]   = 0.0f;
        sumsq[i] = 0.0f;
        mn[i]    = INFINITY;
        mx[i]    = -INFINITY;
    }
    if (i < nn) cnt[i] = 0.0f;
}

// monotone float min/max via int/uint atomics (valid with +/-INF init)
__device__ inline void atomicMinF(float* addr, float v) {
    if (v >= 0.0f) atomicMin((int*)addr, __float_as_int(v));
    else           atomicMax((unsigned int*)addr, __float_as_uint(v));
}
__device__ inline void atomicMaxF(float* addr, float v) {
    if (v >= 0.0f) atomicMax((int*)addr, __float_as_int(v));
    else           atomicMin((unsigned int*)addr, __float_as_uint(v));
}

// ------------------------- edge scatter ------------------------------------
// one wave per edge, 4 channels per lane: m = Pd[dst] + Ps[src] (bias folded
// into Pd); accumulate sum / sumsq / min / max / count into L2-resident bufs.
__global__ void edge_scatter_kernel(const long long* __restrict__ ei,
                                    const float* __restrict__ Pd,
                                    const float* __restrict__ Ps,
                                    float* __restrict__ sum,
                                    float* __restrict__ sumsq,
                                    float* __restrict__ mn,
                                    float* __restrict__ mx,
                                    float* __restrict__ cnt)
{
    int t    = blockIdx.x * blockDim.x + threadIdx.x;
    int e    = t >> 5;
    int lane = t & 31;
    if (e >= NEDGES) return;
    long long src = ei[e];
    long long dst = ei[NEDGES + e];

    float4 a = *(const float4*)(Pd + (size_t)dst * FDIM + lane * 4);
    float4 b = *(const float4*)(Ps + (size_t)src * FDIM + lane * 4);
    float m0 = a.x + b.x, m1 = a.y + b.y, m2 = a.z + b.z, m3 = a.w + b.w;

    size_t o = (size_t)dst * FDIM + lane * 4;
    atomicAdd(sum + o + 0, m0);       atomicAdd(sum + o + 1, m1);
    atomicAdd(sum + o + 2, m2);       atomicAdd(sum + o + 3, m3);
    atomicAdd(sumsq + o + 0, m0*m0);  atomicAdd(sumsq + o + 1, m1*m1);
    atomicAdd(sumsq + o + 2, m2*m2);  atomicAdd(sumsq + o + 3, m3*m3);
    atomicMinF(mn + o + 0, m0); atomicMinF(mn + o + 1, m1);
    atomicMinF(mn + o + 2, m2); atomicMinF(mn + o + 3, m3);
    atomicMaxF(mx + o + 0, m0); atomicMaxF(mx + o + 1, m1);
    atomicMaxF(mx + o + 2, m2); atomicMaxF(mx + o + 3, m3);
    if (lane == 0) atomicAdd(cnt + dst, 1.0f);
}

// ------------------------- aggregation finalize ----------------------------
// A16[n] = [mean | sum | std | min | max]  (f16, 640 wide), plus the two
// per-node degree scalers s1 = logd/avg, s2 = avg/logd.
__global__ void finalize_agg_kernel(const float* __restrict__ cnt,
                                    const float* __restrict__ sum,
                                    const float* __restrict__ sumsq,
                                    const float* __restrict__ mn,
                                    const float* __restrict__ mx,
                                    _Float16* __restrict__ A16,
                                    float* __restrict__ s1,
                                    float* __restrict__ s2,
                                    const float* __restrict__ avg_ptr, int nn)
{
    int i = blockIdx.x * blockDim.x + threadIdx.x;
    if (i >= nn * FDIM) return;
    int n = i >> 7, k = i & (FDIM - 1);
    float c    = cnt[n];
    float safe = fmaxf(c, 1.0f);
    float s    = sum[i];
    float mean = s / safe;
    float var  = sumsq[i] / safe - mean * mean;
    if (var < 0.0f) var = 0.0f;
    float sd   = sqrtf(var + 1e-5f);
    float mnv  = (c == 0.0f) ? 0.0f : mn[i];
    float mxv  = (c == 0.0f) ? 0.0f : mx[i];
    _Float16* row = A16 + (size_t)n * (5 * FDIM);
    row[k]            = (_Float16)mean;
    row[FDIM + k]     = (_Float16)s;
    row[2 * FDIM + k] = (_Float16)sd;
    row[3 * FDIM + k] = (_Float16)mnv;
    row[4 * FDIM + k] = (_Float16)mxv;
    if (k == 0) {
        float logd = logf(safe + 1.0f);
        float avg  = *avg_ptr;
        s1[n] = logd / avg;
        s2[n] = avg / logd;
    }
}

// ------------------------- host-side orchestration -------------------------
static inline void run_gemm(hipStream_t st, const _Float16* A, int lda,
                            const _Float16* Bt, int ldk, int r0,
                            float* C, int ldc, const float* bias,
                            const float* rowscale, int M, int K, int Nout,
                            int accum)
{
    int tilesM = M >> 4;                 // M is always a multiple of 16 here
    int tilesN = (Nout + 15) >> 4;
    dim3 grid(tilesM, (tilesN + 3) / 4);
    wmma_gemm_kernel<<<grid, 128, 0, st>>>(A, lda, Bt, ldk, r0, C, ldc,
                                           bias, rowscale, K, Nout, tilesN,
                                           accum);
}

static inline void run_wtrans(hipStream_t st, const float* W, _Float16* Bt,
                              int Kf, int Nc, int Ncpad)
{
    int n = Ncpad * Kf;
    wtrans_f16_kernel<<<(n + 255) / 256, 256, 0, st>>>(W, Bt, Kf, Nc, Ncpad);
}

extern "C" void kernel_launch(void* const* d_in, const int* in_sizes, int n_in,
                              void* d_out, int out_size, void* d_ws, size_t ws_size,
                              hipStream_t stream)
{
    (void)in_sizes; (void)n_in; (void)out_size; (void)ws_size;
    const int N = NNODES;

    const float*     x       = (const float*)d_in[0];
    const long long* ei_in   = (const long long*)d_in[1];
    const long long* ei_out  = (const long long*)d_in[2];
    const float* in_pre_W  = (const float*)d_in[3];
    const float* in_pre_b  = (const float*)d_in[4];
    const float* in_post_W = (const float*)d_in[5];
    const float* in_post_b = (const float*)d_in[6];
    const float* in_lin_W  = (const float*)d_in[7];
    const float* in_lin_b  = (const float*)d_in[8];
    const float* out_pre_W  = (const float*)d_in[9];
    const float* out_pre_b  = (const float*)d_in[10];
    const float* out_post_W = (const float*)d_in[11];
    const float* out_post_b = (const float*)d_in[12];
    const float* out_lin_W  = (const float*)d_in[13];
    const float* out_lin_b  = (const float*)d_in[14];
    const float* comb_W = (const float*)d_in[15];
    const float* comb_b = (const float*)d_in[16];
    const float* head_W = (const float*)d_in[17];
    const float* head_b = (const float*)d_in[18];
    const float* avg_in  = (const float*)d_in[19];
    const float* avg_out = (const float*)d_in[20];

    // ---- workspace carve (256B aligned) ----
    char* p = (char*)d_ws;
    auto carve = [&](size_t bytes) -> char* {
        char* r = p; p += (bytes + 255) & ~(size_t)255; return r;
    };
    _Float16* xh     = (_Float16*)carve((size_t)N * FDIM * 2);
    _Float16* xh2    = (_Float16*)carve((size_t)N * FDIM * 2);
    float*    Pd     = (float*)carve((size_t)N * FDIM * 4);
    float*    Ps     = (float*)carve((size_t)N * FDIM * 4);
    float*    sumb   = (float*)carve((size_t)N * FDIM * 4);
    float*    sumsq  = (float*)carve((size_t)N * FDIM * 4);
    float*    mnb    = (float*)carve((size_t)N * FDIM * 4);
    float*    mxb    = (float*)carve((size_t)N * FDIM * 4);
    float*    cntb   = (float*)carve((size_t)N * 4);
    _Float16* A16    = (_Float16*)carve((size_t)N * 5 * FDIM * 2);
    float*    s1     = (float*)carve((size_t)N * 4);
    float*    s2     = (float*)carve((size_t)N * 4);
    float*    acc64  = (float*)carve((size_t)N * 64 * 4);
    float*    accLin = (float*)carve((size_t)N * 64 * 4);
    _Float16* h64h   = (_Float16*)carve((size_t)N * 64 * 2);
    _Float16* hin16  = (_Float16*)carve((size_t)N * 64 * 2);
    _Float16* hout16 = (_Float16*)carve((size_t)N * 64 * 2);
    float*    acc128 = (float*)carve((size_t)N * FDIM * 4);

    // transposed f16 weights: [layer][dir]
    _Float16* BtPre[NLAYER][2];
    _Float16* BtPost[NLAYER][2];
    _Float16* BtLin[NLAYER][2];
    _Float16* BtComb[NLAYER];
    _Float16* BtHead;
    for (int i = 0; i < NLAYER; ++i)
        for (int d = 0; d < 2; ++d) {
            BtPre[i][d]  = (_Float16*)carve((size_t)128 * 256 * 2);
            BtPost[i][d] = (_Float16*)carve((size_t)64 * 2048 * 2);
            BtLin[i][d]  = (_Float16*)carve((size_t)64 * 64 * 2);
        }
    for (int i = 0; i < NLAYER; ++i)
        BtComb[i] = (_Float16*)carve((size_t)128 * 256 * 2);
    BtHead = (_Float16*)carve((size_t)16 * 128 * 2);

    // ---- convert all weights (transpose, f16, pad columns) ----
    for (int i = 0; i < NLAYER; ++i) {
        run_wtrans(stream, in_pre_W   + (size_t)i * 256 * 128, BtPre[i][0],  256, 128, 128);
        run_wtrans(stream, out_pre_W  + (size_t)i * 256 * 128, BtPre[i][1],  256, 128, 128);
        run_wtrans(stream, in_post_W  + (size_t)i * 2048 * 64, BtPost[i][0], 2048, 64, 64);
        run_wtrans(stream, out_post_W + (size_t)i * 2048 * 64, BtPost[i][1], 2048, 64, 64);
        run_wtrans(stream, in_lin_W   + (size_t)i * 64 * 64,   BtLin[i][0],  64, 64, 64);
        run_wtrans(stream, out_lin_W  + (size_t)i * 64 * 64,   BtLin[i][1],  64, 64, 64);
        run_wtrans(stream, comb_W     + (size_t)i * 256 * 128, BtComb[i],    256, 128, 128);
    }
    run_wtrans(stream, head_W, BtHead, 128, 8, 16);

    const int EB = 256;
    int gNF  = (N * FDIM + EB - 1) / EB;
    int gN64 = (N * 64 + EB - 1) / EB;

    // x (f32) -> f16 activation copy
    f32_to_f16_kernel<<<gNF, EB, 0, stream>>>(x, xh, N * FDIM);

    for (int i = 0; i < NLAYER; ++i) {
        for (int dir = 0; dir < 2; ++dir) {
            const long long* ei = dir ? ei_out : ei_in;
            const float* preB  = (dir ? out_pre_b  : in_pre_b)  + (size_t)i * 128;
            const float* postB = (dir ? out_post_b : in_post_b) + (size_t)i * 64;
            const float* linB  = (dir ? out_lin_b  : in_lin_b)  + (size_t)i * 64;
            const float* avgp  = dir ? avg_out : avg_in;
            _Float16*    hdir  = dir ? hout16 : hin16;

            // factored edge pre-MLP: Pd = X @ W[0:128] + b ; Ps = X @ W[128:256]
            run_gemm(stream, xh, FDIM, BtPre[i][dir], 256, 0,   Pd, FDIM, preB,    nullptr, N, 128, 128, 0);
            run_gemm(stream, xh, FDIM, BtPre[i][dir], 256, 128, Ps, FDIM, nullptr, nullptr, N, 128, 128, 0);

            init_agg_kernel<<<gNF, EB, 0, stream>>>(sumb, sumsq, mnb, mxb, cntb, N);
            int gE = (NEDGES * 32 + EB - 1) / EB;
            edge_scatter_kernel<<<gE, EB, 0, stream>>>(ei, Pd, Ps, sumb, sumsq, mnb, mxb, cntb);
            finalize_agg_kernel<<<gNF, EB, 0, stream>>>(cntb, sumb, sumsq, mnb, mxb,
                                                        A16, s1, s2, avgp, N);

            // post MLP, degree scalers folded into per-row GEMM epilogues:
            // acc64 = X@W[0:128]+b + A@W[128:768] + s1*(A@W[768:1408]) + s2*(A@W[1408:2048])
            run_gemm(stream, xh,  FDIM,   BtPost[i][dir], 2048, 0,    acc64, 64, postB,   nullptr, N, 128, 64, 0);
            run_gemm(stream, A16, 5*FDIM, BtPost[i][dir], 2048, 128,  acc64, 64, nullptr, nullptr, N, 640, 64, 1);
            run_gemm(stream, A16, 5*FDIM, BtPost[i][dir], 2048, 768,  acc64, 64, nullptr, s1,      N, 640, 64, 1);
            run_gemm(stream, A16, 5*FDIM, BtPost[i][dir], 2048, 1408, acc64, 64, nullptr, s2,      N, 640, 64, 1);

            // lin + relu -> f16 per-direction hidden
            f32_to_f16_kernel<<<gN64, EB, 0, stream>>>(acc64, h64h, N * 64);
            run_gemm(stream, h64h, 64, BtLin[i][dir], 64, 0, accLin, 64, linB, nullptr, N, 64, 64, 0);
            relu_f32_to_f16_kernel<<<gN64, EB, 0, stream>>>(accLin, hdir, N * 64);
        }

        // comb: relu(cat([x, h_in, h_out]) @ W + b), factored over W row blocks
        const float* cB = comb_b + (size_t)i * 128;
        run_gemm(stream, xh,     FDIM, BtComb[i], 256, 0,   acc128, FDIM, cB,      nullptr, N, 128, 128, 0);
        run_gemm(stream, hin16,  64,   BtComb[i], 256, 128, acc128, FDIM, nullptr, nullptr, N, 64,  128, 1);
        run_gemm(stream, hout16, 64,   BtComb[i], 256, 192, acc128, FDIM, nullptr, nullptr, N, 64,  128, 1);
        relu_f32_to_f16_kernel<<<gNF, EB, 0, stream>>>(acc128, xh2, N * FDIM);
        _Float16* t = xh; xh = xh2; xh2 = t;
    }

    // head: out[N x 8] = X @ head_W + head_b (padded Bt rows, guarded store)
    run_gemm(stream, xh, FDIM, BtHead, 128, 0, (float*)d_out, 8, head_b,
             nullptr, N, 128, 8, 0);
}
